// SmallTransformerBlock_33809982554526
// MI455X (gfx1250) — compile-verified, matrix operand order checked by reading, not run on hardware
//
#include <hip/hip_runtime.h>

// MI455X / gfx1250 transformer block: all GEMMs via v_wmma_f32_16x16x32_f16
// (f16 operands, fp32 accumulate), fp32 softmax/LayerNorm/bias/residual.
// Round 4: hoisted scalar row-base pointers + integer k indexing (no pointer
// phis) so loads stay on the GLOBAL path (global_load_b128, LOADcnt only)
// instead of FLAT (which also ties up DScnt/LDS arbitration); unroll 2 keeps
// 16 v_wmma per loop body.

typedef __attribute__((ext_vector_type(16))) _Float16 v16h;
typedef __attribute__((ext_vector_type(8)))  _Float16 v8h;
typedef __attribute__((ext_vector_type(4)))  _Float16 v4h;
typedef __attribute__((ext_vector_type(8)))  float    v8f;

static __device__ __forceinline__ v16h cat16(v8h lo, v8h hi) {
  return __builtin_shufflevector(lo, hi, 0, 1, 2, 3, 4, 5, 6, 7,
                                 8, 9, 10, 11, 12, 13, 14, 15);
}

// ---------------------------------------------------------------- cvt f32->f16
__global__ __launch_bounds__(256) void cvt_f32_to_f16_k(
    const float* __restrict__ in, _Float16* __restrict__ out, long n4) {
  long i = (long)blockIdx.x * blockDim.x + threadIdx.x;
  if (i >= n4) return;
  float4 v = reinterpret_cast<const float4*>(in)[i];
  v4h h;
  h.x = (_Float16)v.x; h.y = (_Float16)v.y;
  h.z = (_Float16)v.z; h.w = (_Float16)v.w;
  reinterpret_cast<v4h*>(out)[i] = h;
}

// --------------------------------------------- weight transpose + cvt to f16
// in: [K][N] f32 row-major  ->  out: [N][K] f16 row-major (K-major for WMMA B)
__global__ void transpose_cvt_k(const float* __restrict__ in,
                                _Float16* __restrict__ out, int K, int N) {
  __shared__ _Float16 tile[32][33];
  const int k0 = blockIdx.y * 32, n0 = blockIdx.x * 32;
  const int tx = threadIdx.x, ty = threadIdx.y;
  for (int i = ty; i < 32; i += 8)
    tile[i][tx] = (_Float16)in[(long)(k0 + i) * N + n0 + tx];
  __syncthreads();
  for (int i = ty; i < 32; i += 8)
    out[(long)(n0 + i) * K + k0 + tx] = tile[tx][i];
}

// ----------------------------------------------------------- generic WMMA GEMM
// C[b][m][n] = scale * sum_k A[b][m][k] * Bt[b][n][k]  (+bias +resid, relu)
// Block: 256 threads = 8 waves (4 along M x 2 along N); block tile 128x128;
// wave tile 32(M) x 64(N) = 2x4 fragments -> 8 v_wmma per 32-deep K step.
// Fragment addressing per CDNA5 ISA 16-bit A/B layouts (two 8-half runs at
// k-offsets lh*8 and 16+lh*8, row/col = lane%16).
template <bool BIAS, bool RESID, bool OUTF, bool OUTH, bool OUTHT, bool RELU>
__global__ __launch_bounds__(256) void gemm_wmma_k(
    const _Float16* __restrict__ A, const _Float16* __restrict__ Bt,
    const float* __restrict__ bias, const float* __restrict__ resid,
    float* __restrict__ outF, _Float16* __restrict__ outH,
    _Float16* __restrict__ outHT,
    int N, int K, int Tb, long sA, long sB, long sC, float scale) {
  const int lane = threadIdx.x & 31;
  const int wid  = threadIdx.x >> 5;
  const int lr   = lane & 15;   // row/col within fragment
  const int lh   = lane >> 4;   // lane-half selects k sub-window
  const int bz   = blockIdx.z;
  const int m0 = blockIdx.y * 128 + (wid & 3) * 32;
  const int n0 = blockIdx.x * 128 + (wid >> 2) * 64;

  // Loop-invariant row-base pointers, derived directly from kernel args
  // (scalars, no pointer phis -> address-space inference keeps loads GLOBAL).
  const _Float16* pa0 = A  + (long)bz * sA + (long)(m0 + lr) * K + lh * 8;
  const _Float16* pa1 = pa0 + 16 * (long)K;
  const _Float16* pb0 = Bt + (long)bz * sB + (long)(n0 + lr) * K + lh * 8;
  const _Float16* pb1 = pb0 + 16 * (long)K;
  const _Float16* pb2 = pb0 + 32 * (long)K;
  const _Float16* pb3 = pb0 + 48 * (long)K;

#define LDF(p, k) cat16(*reinterpret_cast<const v8h*>((p) + (k)), \
                        *reinterpret_cast<const v8h*>((p) + (k) + 16))

  v8f acc[2][4] = {};
#pragma unroll 2
  for (int k = 0; k < K; k += 32) {
    v16h a0 = LDF(pa0, k);
    v16h a1 = LDF(pa1, k);
    v16h b0 = LDF(pb0, k);
    v16h b1 = LDF(pb1, k);
    v16h b2 = LDF(pb2, k);
    v16h b3 = LDF(pb3, k);
    acc[0][0] = __builtin_amdgcn_wmma_f32_16x16x32_f16(
        false, a0, false, b0, (short)0, acc[0][0], false, false);
    acc[0][1] = __builtin_amdgcn_wmma_f32_16x16x32_f16(
        false, a0, false, b1, (short)0, acc[0][1], false, false);
    acc[0][2] = __builtin_amdgcn_wmma_f32_16x16x32_f16(
        false, a0, false, b2, (short)0, acc[0][2], false, false);
    acc[0][3] = __builtin_amdgcn_wmma_f32_16x16x32_f16(
        false, a0, false, b3, (short)0, acc[0][3], false, false);
    acc[1][0] = __builtin_amdgcn_wmma_f32_16x16x32_f16(
        false, a1, false, b0, (short)0, acc[1][0], false, false);
    acc[1][1] = __builtin_amdgcn_wmma_f32_16x16x32_f16(
        false, a1, false, b1, (short)0, acc[1][1], false, false);
    acc[1][2] = __builtin_amdgcn_wmma_f32_16x16x32_f16(
        false, a1, false, b2, (short)0, acc[1][2], false, false);
    acc[1][3] = __builtin_amdgcn_wmma_f32_16x16x32_f16(
        false, a1, false, b3, (short)0, acc[1][3], false, false);
  }
#undef LDF

  // Epilogue: D layout -> element (m0+i*16+lh*8+r, n0+j*16+lr) in VGPR r.
#pragma unroll
  for (int i = 0; i < 2; ++i) {
#pragma unroll
    for (int j = 0; j < 4; ++j) {
      const int n = n0 + j * 16 + lr;
      float bv = 0.0f;
      if (BIAS) bv = bias[n];
#pragma unroll
      for (int r = 0; r < 8; ++r) {
        const int m = m0 + i * 16 + lh * 8 + r;
        float v = acc[i][j][r] * scale + bv;
        const long idx = (long)bz * sC + (long)m * N + n;
        if (RESID) v += resid[idx];
        if (RELU) v = fmaxf(v, 0.0f);
        if (OUTF) outF[idx] = v;
        if (OUTH) outH[idx] = (_Float16)v;
        if (OUTHT) {  // per-batch transposed f16 store: [b][n][t]
          const int bb = m / Tb, t = m - bb * Tb;
          outHT[((long)bb * N + n) * Tb + t] = (_Float16)v;
        }
      }
    }
  }
}

// ----------------------------------------------------------- row softmax 2048
__global__ __launch_bounds__(256) void softmax_k(
    const float* __restrict__ s, _Float16* __restrict__ attn) {
  __shared__ float red[256];
  const int tid = threadIdx.x;
  const long base = (long)blockIdx.x * 2048;
  float loc[8];
  float mx = -3.4e38f;
#pragma unroll
  for (int i = 0; i < 8; ++i) {
    loc[i] = s[base + tid + i * 256];
    mx = fmaxf(mx, loc[i]);
  }
  red[tid] = mx; __syncthreads();
  for (int st = 128; st > 0; st >>= 1) {
    if (tid < st) red[tid] = fmaxf(red[tid], red[tid + st]);
    __syncthreads();
  }
  mx = red[0]; __syncthreads();
  float sum = 0.0f;
#pragma unroll
  for (int i = 0; i < 8; ++i) { loc[i] = __expf(loc[i] - mx); sum += loc[i]; }
  red[tid] = sum; __syncthreads();
  for (int st = 128; st > 0; st >>= 1) {
    if (tid < st) red[tid] += red[tid + st];
    __syncthreads();
  }
  const float inv = 1.0f / red[0];
#pragma unroll
  for (int i = 0; i < 8; ++i)
    attn[base + tid + i * 256] = (_Float16)(loc[i] * inv);
}

// ----------------------------------------------------------- LayerNorm (1024)
__global__ __launch_bounds__(256) void layernorm_k(
    const float* __restrict__ in, const float* __restrict__ g,
    const float* __restrict__ be, float* __restrict__ outF,
    _Float16* __restrict__ outH) {
  __shared__ float red[256];
  const int tid = threadIdx.x;
  const long base = (long)blockIdx.x * 1024;
  float loc[4];
  float s = 0.0f;
#pragma unroll
  for (int i = 0; i < 4; ++i) { loc[i] = in[base + tid + i * 256]; s += loc[i]; }
  red[tid] = s; __syncthreads();
  for (int st = 128; st > 0; st >>= 1) {
    if (tid < st) red[tid] += red[tid + st];
    __syncthreads();
  }
  const float mean = red[0] * (1.0f / 1024.0f); __syncthreads();
  float v = 0.0f;
#pragma unroll
  for (int i = 0; i < 4; ++i) { const float d = loc[i] - mean; v += d * d; }
  red[tid] = v; __syncthreads();
  for (int st = 128; st > 0; st >>= 1) {
    if (tid < st) red[tid] += red[tid + st];
    __syncthreads();
  }
  const float inv = rsqrtf(red[0] * (1.0f / 1024.0f) + 1e-5f);
#pragma unroll
  for (int i = 0; i < 4; ++i) {
    const int c = tid + i * 256;
    const float y = (loc[i] - mean) * inv * g[c] + be[c];
    if (outF) outF[base + c] = y;
    if (outH) outH[base + c] = (_Float16)y;
  }
}

// ---------------------------------------------------------------------- driver
extern "C" void kernel_launch(void* const* d_in, const int* in_sizes, int n_in,
                              void* d_out, int out_size, void* d_ws,
                              size_t ws_size, hipStream_t stream) {
  (void)in_sizes; (void)n_in; (void)out_size; (void)ws_size;
  const int D = 1024, F = 4096, NB = 4, T = 2048;
  const long M = (long)NB * T;  // 8192

  const float* x     = (const float*)d_in[0];
  const float* wq    = (const float*)d_in[1];
  const float* bq    = (const float*)d_in[2];
  const float* wk    = (const float*)d_in[3];
  const float* bk    = (const float*)d_in[4];
  const float* wv    = (const float*)d_in[5];
  const float* bv    = (const float*)d_in[6];
  const float* wo    = (const float*)d_in[7];
  const float* bo    = (const float*)d_in[8];
  const float* w1    = (const float*)d_in[9];
  const float* b1    = (const float*)d_in[10];
  const float* w2    = (const float*)d_in[11];
  const float* b2    = (const float*)d_in[12];
  const float* g1    = (const float*)d_in[13];
  const float* beta1 = (const float*)d_in[14];
  const float* g2    = (const float*)d_in[15];
  const float* beta2 = (const float*)d_in[16];

  char* ws = (char*)d_ws;
  size_t cur = 0;
  auto alloc = [&](size_t bytes) -> void* {
    void* p = ws + cur;
    cur += (bytes + 255) & ~(size_t)255;
    return p;
  };

  _Float16* x16    = (_Float16*)alloc(M * D * 2);
  _Float16* wqT    = (_Float16*)alloc((long)D * D * 2);
  _Float16* wkT    = (_Float16*)alloc((long)D * D * 2);
  _Float16* wvT    = (_Float16*)alloc((long)D * D * 2);
  _Float16* woT    = (_Float16*)alloc((long)D * D * 2);
  _Float16* w1T    = (_Float16*)alloc((long)F * D * 2);
  _Float16* w2T    = (_Float16*)alloc((long)D * F * 2);
  _Float16* q16    = (_Float16*)alloc(M * D * 2);
  _Float16* k16    = (_Float16*)alloc(M * D * 2);
  _Float16* v16t   = (_Float16*)alloc(M * D * 2);          // [b][d][t]
  float*    scores = (float*)alloc((long)NB * T * T * 4);  // reused as h16
  _Float16* attn16 = (_Float16*)alloc((long)NB * T * T * 2);
  float*    xattn  = (float*)alloc(M * D * 4);             // reused as out2
  float*    x1     = (float*)alloc(M * D * 4);
  _Float16* ctx16  = q16;                // Q consumed by scores GEMM
  _Float16* x1_16  = k16;                // K consumed by scores GEMM
  _Float16* h16    = (_Float16*)scores;  // scores consumed by softmax
  float*    out2   = xattn;              // xattn consumed by LN1

  const dim3 blk(256);
  const float qk_scale = 1.0f / sqrtf((float)(D / 8));  // 1/sqrt(128)

  // x -> f16
  cvt_f32_to_f16_k<<<dim3((M * D / 4 + 255) / 256), blk, 0, stream>>>(
      x, x16, M * D / 4);

  // weights -> K-major f16
  const dim3 tb(32, 8);
  transpose_cvt_k<<<dim3(D / 32, D / 32), tb, 0, stream>>>(wq, wqT, D, D);
  transpose_cvt_k<<<dim3(D / 32, D / 32), tb, 0, stream>>>(wk, wkT, D, D);
  transpose_cvt_k<<<dim3(D / 32, D / 32), tb, 0, stream>>>(wv, wvT, D, D);
  transpose_cvt_k<<<dim3(D / 32, D / 32), tb, 0, stream>>>(wo, woT, D, D);
  transpose_cvt_k<<<dim3(F / 32, D / 32), tb, 0, stream>>>(w1, w1T, D, F);
  transpose_cvt_k<<<dim3(D / 32, F / 32), tb, 0, stream>>>(w2, w2T, F, D);

  // Q, K (natural f16), V (stored transposed per batch)
  gemm_wmma_k<true, false, false, true, false, false>
      <<<dim3(D / 128, M / 128, 1), blk, 0, stream>>>(
          x16, wqT, bq, nullptr, nullptr, q16, nullptr, D, D, 1, 0, 0, 0, 1.0f);
  gemm_wmma_k<true, false, false, true, false, false>
      <<<dim3(D / 128, M / 128, 1), blk, 0, stream>>>(
          x16, wkT, bk, nullptr, nullptr, k16, nullptr, D, D, 1, 0, 0, 0, 1.0f);
  gemm_wmma_k<true, false, false, false, true, false>
      <<<dim3(D / 128, M / 128, 1), blk, 0, stream>>>(
          x16, wvT, bv, nullptr, nullptr, nullptr, v16t, D, D, T, 0, 0, 0, 1.0f);

  // scores[b] = Q[b] @ K[b]^T / sqrt(128)  (K natural layout == WMMA B layout)
  gemm_wmma_k<false, false, true, false, false, false>
      <<<dim3(T / 128, T / 128, NB), blk, 0, stream>>>(
          q16, k16, nullptr, nullptr, scores, nullptr, nullptr, T, D, 1,
          (long)T * D, (long)T * D, (long)T * T, qk_scale);

  // softmax rows -> f16 attn
  softmax_k<<<dim3((unsigned)M), blk, 0, stream>>>(scores, attn16);

  // ctx[b] = attn[b] @ V[b]   (B from V^T layout)
  gemm_wmma_k<false, false, false, true, false, false>
      <<<dim3(D / 128, T / 128, NB), blk, 0, stream>>>(
          attn16, v16t, nullptr, nullptr, nullptr, ctx16, nullptr, D, T, 1,
          (long)T * T, (long)D * T, (long)T * D, 1.0f);

  // xattn = x + ctx @ wo + bo
  gemm_wmma_k<true, true, true, false, false, false>
      <<<dim3(D / 128, M / 128, 1), blk, 0, stream>>>(
          ctx16, woT, bo, x, xattn, nullptr, nullptr, D, D, 1, 0, 0, 0, 1.0f);

  // LN1 -> x1 (f32) + x1_16 (f16)
  layernorm_k<<<dim3((unsigned)M), blk, 0, stream>>>(xattn, g1, beta1, x1, x1_16);

  // h = relu(x1 @ w1 + b1) -> f16
  gemm_wmma_k<true, false, false, true, false, true>
      <<<dim3(F / 128, M / 128, 1), blk, 0, stream>>>(
          x1_16, w1T, b1, nullptr, nullptr, h16, nullptr, F, D, 1, 0, 0, 0, 1.0f);

  // out2 = x1 + h @ w2 + b2
  gemm_wmma_k<true, true, true, false, false, false>
      <<<dim3(D / 128, M / 128, 1), blk, 0, stream>>>(
          h16, w2T, b2, x1, out2, nullptr, nullptr, D, F, 1, 0, 0, 0, 1.0f);

  // LN2 -> d_out
  layernorm_k<<<dim3((unsigned)M), blk, 0, stream>>>(out2, g2, beta2,
                                                     (float*)d_out, nullptr);
}
